// MaskQueryDecoder_66812511256727
// MI455X (gfx1250) — matrix operand order repaired and checked by reading, not compiled, not run on hardware
//
#include <hip/hip_runtime.h>
#include <hip/hip_bf16.h>
#include <cmath>

#define C_DIM 256
#define H_DIM 8
#define D_DIM 32
#define B_SEG 8
#define QPER  128
#define KVPER 1024
#define NQ    (B_SEG * QPER)    // 1024
#define NKV   (B_SEG * KVPER)   // 8192

typedef __attribute__((ext_vector_type(16))) _Float16 v16h;
typedef __attribute__((ext_vector_type(8)))  float    v8f;

// ---------------------------------------------------------------------------
// WMMA fragment helpers (contiguous k order k = 16*(lane/16) + s, wave32)
// ---------------------------------------------------------------------------

// A-matrix 16x32 f16 fragment, loaded from an f32 row-major matrix (convert in regs).
// lanes 0-15 : M=lane,    K = k0+{0..15}
// lanes 16-31: M=lane-16, K = k0+{16..31}
__device__ __forceinline__ v16h frag_a_f32(const float* __restrict__ A, int lda,
                                           int row0, int k0, int lane) {
  int m  = row0 + (lane & 15);
  int kb = k0 + ((lane >> 4) << 4);   // +16 for upper half-wave
  const float4* p0 = reinterpret_cast<const float4*>(A + (size_t)m * lda + kb);
  float4 x0 = p0[0], x1 = p0[1], x2 = p0[2], x3 = p0[3];
  v16h a;
  a[0]  = (_Float16)x0.x; a[1]  = (_Float16)x0.y; a[2]  = (_Float16)x0.z; a[3]  = (_Float16)x0.w;
  a[4]  = (_Float16)x1.x; a[5]  = (_Float16)x1.y; a[6]  = (_Float16)x1.z; a[7]  = (_Float16)x1.w;
  a[8]  = (_Float16)x2.x; a[9]  = (_Float16)x2.y; a[10] = (_Float16)x2.z; a[11] = (_Float16)x2.w;
  a[12] = (_Float16)x3.x; a[13] = (_Float16)x3.y; a[14] = (_Float16)x3.z; a[15] = (_Float16)x3.w;
  return a;
}

// B-matrix 32x16 f16 fragment from a K-contiguous f16 matrix (row n holds K values).
// lanes 0-15 : N=lane,    K = k0+{0..15}
// lanes 16-31: N=lane-16, K = k0+{16..31}
// -> 16 contiguous halves per lane = one 32B vector load.
__device__ __forceinline__ v16h frag_b_f16(const _Float16* __restrict__ B, int ldb,
                                           int col0, int k0, int lane) {
  int n  = col0 + (lane & 15);
  int kb = k0 + ((lane >> 4) << 4);   // +16 for upper half-wave
  return *reinterpret_cast<const v16h*>(B + (size_t)n * ldb + kb);
}

__device__ __forceinline__ float gelu_exact(float x) {
  return 0.5f * x * (1.0f + erff(x * 0.70710678118654752440f));
}

// ---------------------------------------------------------------------------
// Generic WMMA GEMM: Y[M,N] = act(A[M,K] @ B^T + bias) (+ residual)
//   A: f32 row-major; B: f16 row-major [N,K] (i.e. the torch weight itself)
//   flags bit0: exact GELU
// One wave -> one 16x16 tile. block = (32,4) -> 4 column tiles per block.
// ---------------------------------------------------------------------------
__global__ void k_gemm_wmma(const float* __restrict__ A, const _Float16* __restrict__ B,
                            const float* __restrict__ bias, const float* __restrict__ resid,
                            float* __restrict__ Y, int M, int N, int K, int flags) {
  int lane = threadIdx.x;
  int ct   = blockIdx.x * blockDim.y + threadIdx.y;
  int col0 = ct << 4;
  if (col0 >= N) return;
  int row0 = blockIdx.y << 4;

  v8f acc = {};
  for (int k0 = 0; k0 < K; k0 += 32) {
    v16h a = frag_a_f32(A, K, row0, k0, lane);
    v16h b = frag_b_f16(B, K, col0, k0, lane);
    acc = __builtin_amdgcn_wmma_f32_16x16x32_f16(false, a, false, b, (short)0, acc,
                                                 false, false);
  }

  int n  = col0 + (lane & 15);
  int mb = row0 + ((lane >> 4) << 3);
  float bv = bias ? bias[n] : 0.0f;
#pragma unroll
  for (int r = 0; r < 8; ++r) {
    float v = acc[r] + bv;
    if (flags & 1) v = gelu_exact(v);
    size_t idx = (size_t)(mb + r) * N + n;
    if (resid) v += resid[idx];
    Y[idx] = v;
  }
}

// ---------------------------------------------------------------------------
// Attention scores: S[bh, i, j] = scale * Q[h, b*qper+i, :] . K[h, b*kper+j, :]
//                   (+ mask[b, i, j] if mask != null).  D=32 -> single WMMA.
// Qp: f32 [H, nq_tot, 32]; Kp: f16 [H, nk_tot, 32]; block = (32,4)
// ---------------------------------------------------------------------------
__global__ void k_scores_wmma(const float* __restrict__ Qp, const _Float16* __restrict__ Kp,
                              const float* __restrict__ mask, float* __restrict__ S,
                              int qper, int kper, int nq_tot, int nk_tot, float scale) {
  int lane = threadIdx.x;
  int jt   = blockIdx.x * blockDim.y + threadIdx.y;
  if ((jt << 4) >= kper) return;
  int it = blockIdx.y;
  int bh = blockIdx.z;
  int b  = bh / H_DIM;
  int h  = bh % H_DIM;

  const float*    Aq = Qp + ((size_t)h * nq_tot + (size_t)b * qper) * D_DIM;
  const _Float16* Bk = Kp + (size_t)h * nk_tot * D_DIM;

  v16h a = frag_a_f32(Aq, D_DIM, it << 4, 0, lane);
  v16h f = frag_b_f16(Bk, D_DIM, b * kper + (jt << 4), 0, lane);
  v8f acc = {};
  acc = __builtin_amdgcn_wmma_f32_16x16x32_f16(false, a, false, f, (short)0, acc,
                                               false, false);

  int j  = (jt << 4) + (lane & 15);
  int ib = (it << 4) + ((lane >> 4) << 3);
  float*       Sb = S + (size_t)bh * qper * kper;
  const float* Mb = mask ? (mask + (size_t)b * qper * kper) : nullptr;
#pragma unroll
  for (int r = 0; r < 8; ++r) {
    float v = acc[r] * scale;
    if (Mb) v += Mb[(size_t)(ib + r) * kper + j];
    Sb[(size_t)(ib + r) * kper + j] = v;
  }
}

// ---------------------------------------------------------------------------
// O[b*qper+i, h*32+d] = sum_j P[bh,i,j] * Vt[h, d, b*kper+j]
// P: f32 [B*H, qper, kper]; Vt: f16 [H, 32, nk_tot]; block = (32,2) (two d-tiles)
// ---------------------------------------------------------------------------
__global__ void k_attnout_wmma(const float* __restrict__ P, const _Float16* __restrict__ Vt,
                               float* __restrict__ O, int qper, int kper, int nk_tot) {
  int lane = threadIdx.x;
  int dt   = threadIdx.y;
  int it   = blockIdx.y;
  int bh   = blockIdx.z;
  int b    = bh / H_DIM;
  int h    = bh % H_DIM;

  const float*    Ap = P  + (size_t)bh * qper * kper;
  const _Float16* Bv = Vt + (size_t)h * D_DIM * nk_tot;

  v8f acc = {};
  for (int k0 = 0; k0 < kper; k0 += 32) {
    v16h a = frag_a_f32(Ap, kper, it << 4, k0, lane);
    v16h f = frag_b_f16(Bv, nk_tot, dt << 4, b * kper + k0, lane);
    acc = __builtin_amdgcn_wmma_f32_16x16x32_f16(false, a, false, f, (short)0, acc,
                                                 false, false);
  }

  int d  = (dt << 4) + (lane & 15);
  int ib = (it << 4) + ((lane >> 4) << 3);
#pragma unroll
  for (int r = 0; r < 8; ++r)
    O[(size_t)(b * qper + ib + r) * C_DIM + h * D_DIM + d] = acc[r];
}

// ---------------------------------------------------------------------------
// RMSNorm over rows of 256 (eps = float32 machine eps, per reference)
// ---------------------------------------------------------------------------
__global__ void k_rms_256(const float* __restrict__ X, const float* __restrict__ W,
                          float* __restrict__ Y, int rows) {
  __shared__ float red[256];
  int row = blockIdx.x;
  int t   = threadIdx.x;
  float x = X[(size_t)row * 256 + t];
  red[t] = x * x;
  __syncthreads();
  for (int s = 128; s > 0; s >>= 1) {
    if (t < s) red[t] += red[t + s];
    __syncthreads();
  }
  float r = rsqrtf(red[0] * (1.0f / 256.0f) + 1.1920928955078125e-07f);
  Y[(size_t)row * 256 + t] = x * r * W[t];
}

// Wave-per-row softmax (in place)
__global__ void k_softmax(float* __restrict__ S, int len) {
  int lane = threadIdx.x;
  float* row = S + (size_t)blockIdx.x * len;
  float m = -3.4e38f;
  for (int i = lane; i < len; i += 32) m = fmaxf(m, row[i]);
#pragma unroll
  for (int o = 16; o > 0; o >>= 1) m = fmaxf(m, __shfl_xor(m, o, 32));
  float s = 0.0f;
  for (int i = lane; i < len; i += 32) {
    float e = expf(row[i] - m);
    row[i] = e;
    s += e;
  }
#pragma unroll
  for (int o = 16; o > 0; o >>= 1) s += __shfl_xor(s, o, 32);
  float inv = 1.0f / s;
  for (int i = lane; i < len; i += 32) row[i] *= inv;
}

__global__ void k_f32_to_f16(const float* __restrict__ src, _Float16* __restrict__ dst, int n) {
  int i = blockIdx.x * blockDim.x + threadIdx.x;
  if (i < n) dst[i] = (_Float16)src[i];
}

// z -> log(sigmoid(z) + 1e-6), in place
__global__ void k_mask_from_z(float* __restrict__ Z, int n) {
  int i = blockIdx.x * blockDim.x + threadIdx.x;
  if (i < n) {
    float p = 1.0f / (1.0f + expf(-Z[i]));
    Z[i] = logf(p + 1e-6f);
  }
}

// pack f32 [M, ldsrc] cols [coloff, coloff+256) into f16 heads:
//   transpose==0: dst[h, m, d]  (K layout, K-contiguous along d)
//   transpose==1: dst[h, d, m]  (V^T layout, K-contiguous along m)
__global__ void k_pack_f16(const float* __restrict__ src, _Float16* __restrict__ dst,
                           int M, int ldsrc, int coloff, int transpose) {
  int i = blockIdx.x * blockDim.x + threadIdx.x;
  if (i >= M * 256) return;
  int m = i >> 8, c = i & 255, h = c >> 5, d = c & 31;
  float v = src[(size_t)m * ldsrc + coloff + c];
  if (transpose) dst[((size_t)h * D_DIM + d) * M + m] = (_Float16)v;
  else           dst[((size_t)h * M + m) * D_DIM + d] = (_Float16)v;
}

// pack f32 [M,256] into f32 [H, M, 32]
__global__ void k_pack_qf32(const float* __restrict__ src, float* __restrict__ dst, int M) {
  int i = blockIdx.x * blockDim.x + threadIdx.x;
  if (i >= M * 256) return;
  int m = i >> 8, c = i & 255, h = c >> 5, d = c & 31;
  dst[((size_t)h * M + m) * D_DIM + d] = src[(size_t)m * 256 + c];
}

// ---------------------------------------------------------------------------
// Host orchestration
// ---------------------------------------------------------------------------
extern "C" void kernel_launch(void* const* d_in, const int* in_sizes, int n_in,
                              void* d_out, int out_size, void* d_ws, size_t ws_size,
                              hipStream_t stream) {
  const float* q        = (const float*)d_in[0];
  const float* kv       = (const float*)d_in[1];
  const float* w_norm_kv= (const float*)d_in[4];
  const float* w_norm1  = (const float*)d_in[5];
  const float* w_norm2  = (const float*)d_in[6];
  const float* w_norm3  = (const float*)d_in[7];
  const float* wq_c     = (const float*)d_in[8];
  const float* wk_c     = (const float*)d_in[9];
  const float* wv_c     = (const float*)d_in[10];
  const float* wo_c     = (const float*)d_in[11];
  const float* bo_c     = (const float*)d_in[12];
  const float* wq_s     = (const float*)d_in[13];
  const float* wkv_s    = (const float*)d_in[14];
  const float* wo_s     = (const float*)d_in[15];
  const float* bo_s     = (const float*)d_in[16];
  const float* w1_mlp   = (const float*)d_in[17];
  const float* b1_mlp   = (const float*)d_in[18];
  const float* w2_mlp   = (const float*)d_in[19];
  const float* b2_mlp   = (const float*)d_in[20];
  const float* w1_mask  = (const float*)d_in[21];
  const float* b1_mask  = (const float*)d_in[22];
  const float* w2_mask  = (const float*)d_in[23];
  const float* b2_mask  = (const float*)d_in[24];
  float* out = (float*)d_out;

  size_t off = 0;
  auto alloc = [&](size_t bytes) -> char* {
    char* p = (char*)d_ws + off;
    off += (bytes + 255) & ~(size_t)255;
    return p;
  };

  float*    kvn      = (float*)   alloc((size_t)NKV * C_DIM * 4);
  _Float16* kvn_h    = (_Float16*)alloc((size_t)NKV * C_DIM * 2);
  _Float16* wq_c_h   = (_Float16*)alloc(C_DIM * C_DIM * 2);
  _Float16* wk_c_h   = (_Float16*)alloc(C_DIM * C_DIM * 2);
  _Float16* wv_c_h   = (_Float16*)alloc(C_DIM * C_DIM * 2);
  _Float16* wo_c_h   = (_Float16*)alloc(C_DIM * C_DIM * 2);
  _Float16* wq_s_h   = (_Float16*)alloc(C_DIM * C_DIM * 2);
  _Float16* wkv_s_h  = (_Float16*)alloc(2 * C_DIM * C_DIM * 2);
  _Float16* wo_s_h   = (_Float16*)alloc(C_DIM * C_DIM * 2);
  _Float16* w1_mlp_h = (_Float16*)alloc(4 * C_DIM * C_DIM * 2);
  _Float16* w2_mlp_h = (_Float16*)alloc(C_DIM * 4 * C_DIM * 2);
  _Float16* w1_mask_h= (_Float16*)alloc(C_DIM * C_DIM * 2);
  _Float16* w2_mask_h= (_Float16*)alloc(C_DIM * C_DIM * 2);
  float*    mask_h   = (float*)   alloc((size_t)NQ * C_DIM * 4);
  float*    m_k      = (float*)   alloc((size_t)NQ * C_DIM * 4);
  float*    amask    = (float*)   alloc((size_t)B_SEG * QPER * KVPER * 4);
  float*    qn       = (float*)   alloc((size_t)NQ * C_DIM * 4);
  float*    proj_tmp = (float*)   alloc((size_t)NKV * C_DIM * 4);
  float*    qproj    = (float*)   alloc((size_t)NQ * C_DIM * 4);
  float*    qpack    = (float*)   alloc((size_t)H_DIM * NQ * D_DIM * 4);
  _Float16* kpack    = (_Float16*)alloc((size_t)H_DIM * NKV * D_DIM * 2);
  _Float16* vtpack   = (_Float16*)alloc((size_t)H_DIM * D_DIM * NKV * 2);
  _Float16* kpack_s  = (_Float16*)alloc((size_t)H_DIM * NQ * D_DIM * 2);
  _Float16* vtpack_s = (_Float16*)alloc((size_t)H_DIM * D_DIM * NQ * 2);
  float*    kvs      = (float*)   alloc((size_t)NQ * 2 * C_DIM * 4);
  float*    S        = (float*)   alloc((size_t)B_SEG * H_DIM * QPER * KVPER * 4);
  float*    feat     = (float*)   alloc((size_t)NQ * C_DIM * 4);
  float*    q1       = (float*)   alloc((size_t)NQ * C_DIM * 4);
  float*    q2       = (float*)   alloc((size_t)NQ * C_DIM * 4);
  float*    mlph     = (float*)   alloc((size_t)NQ * 4 * C_DIM * 4);
  (void)ws_size; (void)in_sizes; (void)n_in; (void)out_size;

  const float scale = 0.17677669529663687f;  // 1/sqrt(32)
  dim3 w32(32, 1), w32x4(32, 4), w32x2(32, 2);

  auto cvt = [&](const float* s, _Float16* d, int n) {
    k_f32_to_f16<<<dim3((n + 255) / 256), dim3(256), 0, stream>>>(s, d, n);
  };
  auto gemm = [&](const float* A, const _Float16* Bm, const float* bias,
                  const float* resid, float* Y, int M, int N, int K, int flags) {
    k_gemm_wmma<<<dim3((N / 16 + 3) / 4, M / 16), w32x4, 0, stream>>>(
        A, Bm, bias, resid, Y, M, N, K, flags);
  };

  // weight conversions
  cvt(wq_c, wq_c_h, C_DIM * C_DIM);       cvt(wk_c, wk_c_h, C_DIM * C_DIM);
  cvt(wv_c, wv_c_h, C_DIM * C_DIM);       cvt(wo_c, wo_c_h, C_DIM * C_DIM);
  cvt(wq_s, wq_s_h, C_DIM * C_DIM);       cvt(wkv_s, wkv_s_h, 2 * C_DIM * C_DIM);
  cvt(wo_s, wo_s_h, C_DIM * C_DIM);       cvt(w1_mlp, w1_mlp_h, 4 * C_DIM * C_DIM);
  cvt(w2_mlp, w2_mlp_h, 4 * C_DIM * C_DIM);
  cvt(w1_mask, w1_mask_h, C_DIM * C_DIM); cvt(w2_mask, w2_mask_h, C_DIM * C_DIM);

  // kv_n = rms(kv) * w_norm_kv
  k_rms_256<<<dim3(NKV), dim3(256), 0, stream>>>(kv, w_norm_kv, kvn, NKV);
  cvt(kvn, kvn_h, NKV * C_DIM);

  // mask MLP on raw q
  gemm(q, w1_mask_h, b1_mask, nullptr, mask_h, NQ, C_DIM, C_DIM, 1);
  gemm(mask_h, w2_mask_h, b2_mask, nullptr, m_k, NQ, C_DIM, C_DIM, 0);

  // z = m_k @ kv_n^T, block-diagonal segments only
  for (int b = 0; b < B_SEG; ++b)
    gemm(m_k + (size_t)b * QPER * C_DIM, kvn_h + (size_t)b * KVPER * C_DIM,
         nullptr, nullptr, amask + (size_t)b * QPER * KVPER, QPER, KVPER, C_DIM, 0);
  k_mask_from_z<<<dim3((B_SEG * QPER * KVPER + 255) / 256), dim3(256), 0, stream>>>(
      amask, B_SEG * QPER * KVPER);

  // ---- cross attention ----
  k_rms_256<<<dim3(NQ), dim3(256), 0, stream>>>(q, w_norm1, qn, NQ);
  gemm(qn, wq_c_h, nullptr, nullptr, qproj, NQ, C_DIM, C_DIM, 0);
  k_pack_qf32<<<dim3(NQ), dim3(256), 0, stream>>>(qproj, qpack, NQ);
  gemm(kvn, wk_c_h, nullptr, nullptr, proj_tmp, NKV, C_DIM, C_DIM, 0);
  k_pack_f16<<<dim3(NKV), dim3(256), 0, stream>>>(proj_tmp, kpack, NKV, C_DIM, 0, 0);
  gemm(kvn, wv_c_h, nullptr, nullptr, proj_tmp, NKV, C_DIM, C_DIM, 0);
  k_pack_f16<<<dim3(NKV), dim3(256), 0, stream>>>(proj_tmp, vtpack, NKV, C_DIM, 0, 1);

  k_scores_wmma<<<dim3(KVPER / 64, QPER / 16, B_SEG * H_DIM), w32x4, 0, stream>>>(
      qpack, kpack, amask, S, QPER, KVPER, NQ, NKV, scale);
  k_softmax<<<dim3(B_SEG * H_DIM * QPER), w32, 0, stream>>>(S, KVPER);
  k_attnout_wmma<<<dim3(1, QPER / 16, B_SEG * H_DIM), w32x2, 0, stream>>>(
      S, vtpack, feat, QPER, KVPER, NKV);
  gemm(feat, wo_c_h, bo_c, q, q1, NQ, C_DIM, C_DIM, 0);

  // ---- self attention (block-diagonal) ----
  k_rms_256<<<dim3(NQ), dim3(256), 0, stream>>>(q1, w_norm2, qn, NQ);
  gemm(qn, wq_s_h, nullptr, nullptr, qproj, NQ, C_DIM, C_DIM, 0);
  k_pack_qf32<<<dim3(NQ), dim3(256), 0, stream>>>(qproj, qpack, NQ);
  gemm(qn, wkv_s_h, nullptr, nullptr, kvs, NQ, 2 * C_DIM, C_DIM, 0);
  k_pack_f16<<<dim3(NQ), dim3(256), 0, stream>>>(kvs, kpack_s, NQ, 2 * C_DIM, 0, 0);
  k_pack_f16<<<dim3(NQ), dim3(256), 0, stream>>>(kvs, vtpack_s, NQ, 2 * C_DIM, C_DIM, 1);

  k_scores_wmma<<<dim3(QPER / 64, QPER / 16, B_SEG * H_DIM), w32x4, 0, stream>>>(
      qpack, kpack_s, nullptr, S, QPER, QPER, NQ, NQ, scale);
  k_softmax<<<dim3(B_SEG * H_DIM * QPER), w32, 0, stream>>>(S, QPER);
  k_attnout_wmma<<<dim3(1, QPER / 16, B_SEG * H_DIM), w32x2, 0, stream>>>(
      S, vtpack_s, feat, QPER, QPER, NQ);
  gemm(feat, wo_s_h, bo_s, q1, q2, NQ, C_DIM, C_DIM, 0);

  // ---- MLP ----
  k_rms_256<<<dim3(NQ), dim3(256), 0, stream>>>(q2, w_norm3, qn, NQ);
  gemm(qn, w1_mlp_h, b1_mlp, nullptr, mlph, NQ, 4 * C_DIM, C_DIM, 1);
  gemm(mlph, w2_mlp_h, b2_mlp, q2, out, NQ, C_DIM, 4 * C_DIM, 0);
}
